// ContextEncodeInplaceABN_83623013253269
// MI455X (gfx1250) — matrix operand order repaired and checked by reading, not compiled, not run on hardware
//
#include <hip/hip_runtime.h>
#include <math.h>

typedef _Float16 h16v __attribute__((ext_vector_type(16)));
typedef _Float16 h8v  __attribute__((ext_vector_type(8)));
typedef float    f8v  __attribute__((ext_vector_type(8)));

#define LRELU(v) ((v) >= 0.0f ? (v) : 0.01f * (v))

__device__ __forceinline__ h16v cat8(h8v lo, h8v hi) {
  h16v r;
#pragma unroll
  for (int j = 0; j < 8; ++j) { r[j] = lo[j]; r[j + 8] = hi[j]; }
  return r;
}

// ---------------------------------------------------------------------------
// Kernel 0: fold ABN params, convert conv_w / codewords to f16, codeword norms
// ---------------------------------------------------------------------------
__global__ void __launch_bounds__(256) prep_kernel(
    const float* __restrict__ conv_w, const float* __restrict__ codewords,
    const float* __restrict__ pg, const float* __restrict__ pb,
    const float* __restrict__ pm, const float* __restrict__ pv,
    const float* __restrict__ g1, const float* __restrict__ b1,
    const float* __restrict__ m1, const float* __restrict__ v1,
    const float* __restrict__ cb, const float* __restrict__ g2,
    const float* __restrict__ b2, const float* __restrict__ m2,
    const float* __restrict__ v2,
    float* __restrict__ coefPre, float* __restrict__ coefD,
    float* __restrict__ c2w, _Float16* __restrict__ wHalf,
    _Float16* __restrict__ cwHalf) {
  int i = blockIdx.x * 256 + threadIdx.x;
  if (i < 65536) { wHalf[i] = (_Float16)conv_w[i]; return; }
  i -= 65536;
  if (i < 2048) { cwHalf[i] = (_Float16)codewords[i]; return; }
  i -= 2048;
  if (i < 512) {
    float s1 = pg[i] * rsqrtf(pv[i] + 1e-5f);
    float t1 = pb[i] - pm[i] * s1;
    float s2 = g1[i] * rsqrtf(v1[i] + 1e-5f);
    float t2 = b1[i] - m1[i] * s2;
    coefPre[4 * i + 0] = s1; coefPre[4 * i + 1] = t1;
    coefPre[4 * i + 2] = s2; coefPre[4 * i + 3] = t2;
    return;
  }
  i -= 512;
  if (i < 128) {
    float s = g2[i] * rsqrtf(v2[i] + 1e-5f);
    float t = cb[i] * s + b2[i] - m2[i] * s;  // conv bias folded into abn2
    coefD[i] = s; coefD[128 + i] = t;
    return;
  }
  i -= 128;
  if (i < 16) {
    float s = 0.0f;
    for (int d = 0; d < 128; ++d) { float c = codewords[i * 128 + d]; s = fmaf(c, c, s); }
    c2w[i] = s;
  }
}

// ---------------------------------------------------------------------------
// Kernel 1: fused ABN(x)->ABN->1x1conv(WMMA f16)->ABN2 -> X (f16), + spatial SE
// Block: 256 thr = 8 waves; tile = 64 pixels x 128 D; K-loop over C (32/step)
// ---------------------------------------------------------------------------
__global__ void __launch_bounds__(256) gemm_abn_spa_kernel(
    const float* __restrict__ x, const float* __restrict__ coefPre,
    const float* __restrict__ spaw, const _Float16* __restrict__ wHalf,
    const float* __restrict__ coefD, _Float16* __restrict__ Xh,
    float* __restrict__ spa) {
  __shared__ _Float16 yA[64 * 40];    // [pixel][c] padded
  __shared__ _Float16 wB[128 * 40];   // [d][c] padded
  __shared__ float spaPart[256];

  const int t = threadIdx.x;
  const int tile = blockIdx.x;
  const int b = tile >> 6;
  const int hwBase = (tile & 63) << 6;
  const float* xb = x + ((size_t)b * 512) * 4096 + hwBase;

  const int m = t & 63;       // staging pixel (consecutive t -> consecutive m)
  const int cRow = t >> 6;    // staging c group: this thread owns c in [8*cRow, 8*cRow+8)
  const int lane = t & 31;
  const int wave = t >> 5;
  const int g = lane >> 4;
  const int lr = lane & 15;
  const int mw = wave & 3;    // 4 M-tiles of 16 pixels
  const int dw = wave >> 2;   // 2 D-halves of 64
  f8v acc[4] = {};
  float spaAcc = 0.0f;

  for (int kc = 0; kc < 16; ++kc) {
    const int cBase = kc * 32;
    // --- stage activations: x -> pre-ABN -> spa dot -> abn1 -> packed f16 LDS
    {
      h8v yv8;
#pragma unroll
      for (int i = 0; i < 8; ++i) {
        int c_off = cRow * 8 + i;   // 8 contiguous channels per thread
        int c = cBase + c_off;
        float4 cf = ((const float4*)coefPre)[c];
        float xv = xb[(size_t)c * 4096 + m];
        float px = fmaf(xv, cf.x, cf.y); px = LRELU(px);
        spaAcc = fmaf(px, spaw[c], spaAcc);
        float yv = fmaf(px, cf.z, cf.w); yv = LRELU(yv);
        yv8[i] = (_Float16)yv;
      }
      *(h8v*)(&yA[m * 40 + cRow * 8]) = yv8;   // one ds_store_b128
    }
    // --- stage weights (f16, pre-converted)
#pragma unroll
    for (int i2 = 0; i2 < 2; ++i2) {
      int idx8 = t + 256 * i2;
      int d = idx8 >> 2;
      int c8 = (idx8 & 3) << 3;
      h8v wv = *(const h8v*)(wHalf + (size_t)d * 512 + cBase + c8);
      *(h8v*)(&wB[d * 40 + c8]) = wv;
    }
    __syncthreads();
    // --- WMMA: preload A + all 4 B fragments, then run WMMAs back-to-back
    const _Float16* ar = &yA[(mw * 16 + lr) * 40];
    h16v a = cat8(*(const h8v*)(ar + 8 * g), *(const h8v*)(ar + 16 + 8 * g));
    h16v bf[4];
#pragma unroll
    for (int nt = 0; nt < 4; ++nt) {
      const _Float16* br = &wB[(dw * 64 + nt * 16 + lr) * 40];
      bf[nt] = cat8(*(const h8v*)(br + 8 * g), *(const h8v*)(br + 16 + 8 * g));
    }
#pragma unroll
    for (int nt = 0; nt < 4; ++nt) {
      acc[nt] = __builtin_amdgcn_wmma_f32_16x16x32_f16(
          false, a, false, bf[nt], (short)0, acc[nt], false, false);
    }
    __syncthreads();
  }

  // --- epilogue: abn2 (+folded bias) + leaky-relu, store X as f16 (B,N,D)
  const size_t pixBase = (size_t)b * 4096 + hwBase;
#pragma unroll
  for (int nt = 0; nt < 4; ++nt) {
    int d = dw * 64 + nt * 16 + lr;
    float sD = coefD[d];
    float tD = coefD[128 + d];
#pragma unroll
    for (int r = 0; r < 8; ++r) {
      int mm = mw * 16 + 8 * g + r;   // D-layout: vgpr r -> M = r + 8*laneGroup
      float zn = fmaf(acc[nt][r], sD, tD);
      zn = LRELU(zn);
      Xh[(pixBase + mm) * 128 + d] = (_Float16)zn;
    }
  }

  // --- spatial SE: reduce 4 c-group partials per pixel, sigmoid
  spaPart[t] = spaAcc;
  __syncthreads();
  if (t < 64) {
    float s = spaPart[t] + spaPart[t + 64] + spaPart[t + 128] + spaPart[t + 192];
    spa[pixBase + t] = 1.0f / (1.0f + __expf(-s));
  }
}

// ---------------------------------------------------------------------------
// Kernel 2: Encoding — xc via WMMA, softmax over K, A^T X partial aggregation
// Block: 256 thr, 128 pixels of one batch; writes per-block partials (no atomics)
// ---------------------------------------------------------------------------
__global__ void __launch_bounds__(256) encode_kernel(
    const _Float16* __restrict__ Xh, const _Float16* __restrict__ cwHalf,
    const float* __restrict__ scale, const float* __restrict__ c2w,
    float* __restrict__ Epart, float* __restrict__ Apart) {
  __shared__ _Float16 XL[128 * 136];  // [pixel][d] padded
  __shared__ _Float16 cwL[2048];      // [k][d]
  __shared__ float sl2[128 * 17];
  __shared__ _Float16 aL[128 * 16];
  __shared__ float x2[128];
  __shared__ float x2p[2048];

  const int t = threadIdx.x;
  const int blk = blockIdx.x;
  const int b = blk >> 5;
  const int pBase = (blk & 31) << 7;
  const size_t pixBase = (size_t)b * 4096 + pBase;

  ((h8v*)cwL)[t] = ((const h8v*)cwHalf)[t];

  const int dgT = t & 15;
#pragma unroll
  for (int i = 0; i < 8; ++i) {
    int idx8 = t + 256 * i;
    int p = idx8 >> 4;
    h8v xv = *(const h8v*)(Xh + (pixBase + p) * 128 + dgT * 8);
    *(h8v*)(&XL[p * 136 + dgT * 8]) = xv;
    float s = 0.0f;
#pragma unroll
    for (int j = 0; j < 8; ++j) { float f = (float)xv[j]; s = fmaf(f, f, s); }
    x2p[p * 16 + dgT] = s;
  }
  __syncthreads();
  if (t < 128) {
    float s = 0.0f;
#pragma unroll
    for (int q = 0; q < 16; ++q) s += x2p[t * 16 + q];
    x2[t] = s;
  }
  __syncthreads();

  // --- xc = X . C^T via WMMA: wave w owns pixel tile w (16 pixels x 16 k)
  const int lane = t & 31;
  const int wave = t >> 5;
  const int g = lane >> 4;
  const int lr = lane & 15;
  const int mBase = wave * 16;
  const float scl = scale[lr];
  const float c2v = c2w[lr];
  f8v accx = {};
  {
    h16v av[4], bv[4];
#pragma unroll
    for (int kc = 0; kc < 4; ++kc) {
      const _Float16* ar = &XL[(mBase + lr) * 136 + kc * 32];
      av[kc] = cat8(*(const h8v*)(ar + 8 * g), *(const h8v*)(ar + 16 + 8 * g));
      const _Float16* br = &cwL[lr * 128 + kc * 32];
      bv[kc] = cat8(*(const h8v*)(br + 8 * g), *(const h8v*)(br + 16 + 8 * g));
    }
#pragma unroll
    for (int kc = 0; kc < 4; ++kc) {
      accx = __builtin_amdgcn_wmma_f32_16x16x32_f16(
          false, av[kc], false, bv[kc], (short)0, accx, false, false);
    }
  }
#pragma unroll
  for (int r = 0; r < 8; ++r) {
    int p = mBase + 8 * g + r;
    sl2[p * 17 + lr] = scl * (x2[p] - 2.0f * accx[r] + c2v);
  }
  __syncthreads();

  // --- softmax over K per pixel
  if (t < 128) {
    const int p = t;
    float mx = -1e30f;
#pragma unroll
    for (int k = 0; k < 16; ++k) mx = fmaxf(mx, sl2[p * 17 + k]);
    float ev[16]; float sum = 0.0f;
#pragma unroll
    for (int k = 0; k < 16; ++k) { ev[k] = __expf(sl2[p * 17 + k] - mx); sum += ev[k]; }
    float inv = 1.0f / sum;
#pragma unroll
    for (int k = 0; k < 16; ++k) aL[p * 16 + k] = (_Float16)(ev[k] * inv);
  }
  __syncthreads();

  // --- aggregate: thread (k, d-group of 8) accumulates over 128 pixels
  const int k = t >> 4;
  const int dg = t & 15;
  float e[8];
#pragma unroll
  for (int j = 0; j < 8; ++j) e[j] = 0.0f;
  float asum = 0.0f;
  for (int p = 0; p < 128; ++p) {
    float av = (float)aL[p * 16 + k];
    asum += av;
    h8v xv = *(const h8v*)(&XL[p * 136 + dg * 8]);
#pragma unroll
    for (int j = 0; j < 8; ++j) e[j] = fmaf(av, (float)xv[j], e[j]);
  }
  float* eo = Epart + (size_t)blk * 2048 + k * 128 + dg * 8;
#pragma unroll
  for (int j = 0; j < 8; ++j) eo[j] = e[j];
  if (dg == 0) Apart[blk * 16 + k] = asum;
}

// ---------------------------------------------------------------------------
// Kernel 3: deterministic partial reduction, E - Asum*C, L2 normalize -> enc
// ---------------------------------------------------------------------------
__global__ void __launch_bounds__(256) finalize_enc_kernel(
    const float* __restrict__ Epart, const float* __restrict__ Apart,
    const float* __restrict__ codewords, float* __restrict__ encOut) {
  __shared__ float asumL[16];
  __shared__ float red[256];
  const int t = threadIdx.x;
  const int b = blockIdx.x;
  if (t < 16) {
    float s = 0.0f;
    for (int ch = 0; ch < 32; ++ch) s += Apart[(b * 32 + ch) * 16 + t];
    asumL[t] = s;
  }
  __syncthreads();
  float vloc[8];
  float nacc = 0.0f;
#pragma unroll
  for (int i = 0; i < 8; ++i) {
    int idx = t + 256 * i;   // idx = k*128 + d
    float s = 0.0f;
    for (int ch = 0; ch < 32; ++ch) s += Epart[((size_t)(b * 32 + ch)) * 2048 + idx];
    float v = s - asumL[idx >> 7] * codewords[idx];
    vloc[i] = v;
    nacc = fmaf(v, v, nacc);
  }
  red[t] = nacc;
  __syncthreads();
  for (int s2 = 128; s2 > 0; s2 >>= 1) {
    if (t < s2) red[t] += red[t + s2];
    __syncthreads();
  }
  float inv = 1.0f / fmaxf(sqrtf(red[0]), 1e-12f);
#pragma unroll
  for (int i = 0; i < 8; ++i) encOut[b * 2048 + t + 256 * i] = vloc[i] * inv;
}

// ---------------------------------------------------------------------------
// Kernel 4: channel SE — one block per output channel c, all 16 batches
// ---------------------------------------------------------------------------
__global__ void __launch_bounds__(256) chn_se_kernel(
    const float* __restrict__ enc, const float* __restrict__ se_w,
    const float* __restrict__ se_b, float* __restrict__ chn) {
  __shared__ float wrow[2048];
  __shared__ float pp[256];
  const int t = threadIdx.x;
  const int c = blockIdx.x;
#pragma unroll
  for (int i = 0; i < 8; ++i) wrow[t + 256 * i] = se_w[(size_t)c * 2048 + t + 256 * i];
  __syncthreads();
  const int b = t >> 4;
  const int part = t & 15;
  float s = 0.0f;
  for (int j = 0; j < 128; ++j) {
    int idx = part + 16 * j;
    s = fmaf(enc[b * 2048 + idx], wrow[idx], s);
  }
  pp[t] = s;
  __syncthreads();
  if (t < 16) {
    float tot = 0.0f;
#pragma unroll
    for (int q = 0; q < 16; ++q) tot += pp[t * 16 + q];
    tot += se_b[c];
    chn[t * 512 + c] = 1.0f / (1.0f + __expf(-tot));
  }
}

// ---------------------------------------------------------------------------
// Kernel 5: out = x * spa_se * chn_se (float4 elementwise)
// ---------------------------------------------------------------------------
__global__ void __launch_bounds__(256) out_kernel(
    const float* __restrict__ x, const float* __restrict__ spa,
    const float* __restrict__ chn, float* __restrict__ out) {
  const size_t i4 = (size_t)blockIdx.x * 256 + threadIdx.x;
  const float4 xv = ((const float4*)x)[i4];
  const int lin = (int)i4;
  const int hw4 = lin & 1023;
  const int c = (lin >> 10) & 511;
  const int b = lin >> 19;
  const float ch = chn[b * 512 + c];
  const float4 sp = ((const float4*)spa)[b * 1024 + hw4];
  float4 o;
  o.x = xv.x * sp.x * ch; o.y = xv.y * sp.y * ch;
  o.z = xv.z * sp.z * ch; o.w = xv.w * sp.w * ch;
  ((float4*)out)[i4] = o;
}

// ---------------------------------------------------------------------------
extern "C" void kernel_launch(void* const* d_in, const int* in_sizes, int n_in,
                              void* d_out, int out_size, void* d_ws, size_t ws_size,
                              hipStream_t stream) {
  const float* x        = (const float*)d_in[0];
  const float* pg       = (const float*)d_in[1];
  const float* pb       = (const float*)d_in[2];
  const float* pm       = (const float*)d_in[3];
  const float* pv       = (const float*)d_in[4];
  const float* g1       = (const float*)d_in[5];
  const float* b1       = (const float*)d_in[6];
  const float* m1       = (const float*)d_in[7];
  const float* v1       = (const float*)d_in[8];
  const float* conv_w   = (const float*)d_in[9];
  const float* cb       = (const float*)d_in[10];
  const float* g2       = (const float*)d_in[11];
  const float* b2       = (const float*)d_in[12];
  const float* m2       = (const float*)d_in[13];
  const float* v2       = (const float*)d_in[14];
  const float* codewords= (const float*)d_in[15];
  const float* scale    = (const float*)d_in[16];
  const float* se_w     = (const float*)d_in[17];
  const float* se_b     = (const float*)d_in[18];
  const float* spa_w    = (const float*)d_in[19];

  float* enc_out = (float*)d_out;            // 16*2048
  float* out     = (float*)d_out + 32768;    // 16*512*64*64

  // workspace layout (floats first, halves after; half base is 16B aligned)
  float* wsf      = (float*)d_ws;
  float* coefPre  = wsf;                       // 2048
  float* coefD    = coefPre + 2048;            // 256
  float* c2w      = coefD + 256;               // 16
  float* spa      = c2w + 16;                  // 65536
  float* chn      = spa + 65536;               // 8192
  float* Epart    = chn + 8192;                // 16*32*2048 = 1048576
  float* Apart    = Epart + 1048576;           // 16*32*16   = 8192
  _Float16* wHalf  = (_Float16*)(Apart + 8192);// 65536
  _Float16* cwHalf = wHalf + 65536;            // 2048
  _Float16* Xh     = cwHalf + 2048;            // 16*4096*128 = 8388608

  prep_kernel<<<267, 256, 0, stream>>>(conv_w, codewords, pg, pb, pm, pv,
                                       g1, b1, m1, v1, cb, g2, b2, m2, v2,
                                       coefPre, coefD, c2w, wHalf, cwHalf);

  gemm_abn_spa_kernel<<<1024, 256, 0, stream>>>(x, coefPre, spa_w, wHalf,
                                                coefD, Xh, spa);

  encode_kernel<<<512, 256, 0, stream>>>(Xh, cwHalf, scale, c2w, Epart, Apart);

  finalize_enc_kernel<<<16, 256, 0, stream>>>(Epart, Apart, codewords, enc_out);

  chn_se_kernel<<<512, 256, 0, stream>>>(enc_out, se_w, se_b, chn);

  out_kernel<<<32768, 256, 0, stream>>>(x, spa, chn, out);

  (void)in_sizes; (void)n_in; (void)out_size; (void)ws_size;
}
